// AttnGRU_84104049590532
// MI455X (gfx1250) — matrix operand order repaired and checked by reading, not compiled, not run on hardware
//
#include <hip/hip_runtime.h>

// ---------------------------------------------------------------------------
// AttnGRU on MI455X (gfx1250, wave32).  All K>=256 GEMMs run on
// v_wmma_f32_16x16x32_bf16 (bf16 inputs, f32 accumulate).  spat_hidden
// (512 MiB, the dominant traffic) is streamed once per node through LDS.
// ---------------------------------------------------------------------------

#define NN   16384   // nodes
#define EE   32      // edges per node
#define HH   256     // hidden dim
#define AA   64      // attention dim
#define ODIM 5       // output dim

typedef __attribute__((ext_vector_type(16))) __bf16        v16bf;
typedef __attribute__((ext_vector_type(8)))  float         v8f;
typedef __attribute__((ext_vector_type(8)))  unsigned short us8;
typedef __attribute__((ext_vector_type(4)))  unsigned short us4;

// ---- fp32 <-> bf16 (round to nearest even) --------------------------------
static __device__ inline unsigned short f2bf(float f) {
    union { float f; unsigned u; } c; c.f = f;
    unsigned u = c.u;
    unsigned r = u + 0x7fffu + ((u >> 16) & 1u);
    return (unsigned short)(r >> 16);
}
static __device__ inline float bf2f(unsigned short h) {
    union { unsigned u; float f; } c; c.u = ((unsigned)h) << 16;
    return c.f;
}

// ---- WMMA helpers ----------------------------------------------------------
static __device__ inline v8f wmma_bf16(v16bf a, v16bf b, v8f c) {
    // (neg_a, A, neg_b, B, c_mod, C, reuse_a, reuse_b)
    return __builtin_amdgcn_wmma_f32_16x16x32_bf16(false, a, false, b,
                                                   (short)0, c, false, false);
}

// A-fragment, documented 16-bit 16x32 layout:
//   lane<16 : row=lane,     K = {kb..kb+7, kb+16..kb+23}
//   lane>=16: row=lane-16,  K = {kb+8..kb+15, kb+24..kb+31}
static __device__ inline v16bf load_a16(const unsigned short* base, int ld,
                                        int kbase, int lane) {
    int r    = lane & 15;
    int koff = kbase + ((lane & 16) ? 8 : 0);
    const unsigned short* p = base + (size_t)r * ld + koff;
    union { v16bf v; us8 h[2]; } t;
    t.h[0] = *(const us8*)(p);
    t.h[1] = *(const us8*)(p + 16);
    return t.v;
}

// One 16x16 output tile, accumulating over K = ksteps*32.
// Wp is fragment-major packed: tile (nt,ks) -> 32 lanes x 16 bf16.
static __device__ inline v8f gemm_tile(const unsigned short* Xtile, int ldx,
                                       const unsigned short* Wp, int nt,
                                       int ksteps, int lane, v8f acc) {
    const v16bf* bp = (const v16bf*)Wp + (size_t)nt * ksteps * 32 + lane;
    for (int ks = 0; ks < ksteps; ++ks) {
        v16bf a = load_a16(Xtile, ldx, ks * 32, lane);
        v16bf b = bp[(size_t)ks * 32];
        acc = wmma_bf16(a, b, acc);
    }
    return acc;
}

// ---------------------------------------------------------------------------
// Pack W[Aout,K] (row major f32) into bf16 B-fragments.
// dst element index = ((nt*(K/32)+ks)*32 + lane)*16 + i
//   n = nt*16 + (lane&15);  k = ks*32 + ((lane&16)?8:0) + (i&7) + ((i&8)?16:0)
// ---------------------------------------------------------------------------
__global__ void __launch_bounds__(256)
pack_b_kernel(const float* __restrict__ W, unsigned short* __restrict__ dst,
              int Aout, int K) {
    int total = (Aout >> 4) * (K >> 5) * 512;
    int idx = blockIdx.x * 256 + threadIdx.x;
    if (idx >= total) return;
    int i      = idx & 15;
    int lane   = (idx >> 4) & 31;
    int tile   = idx >> 9;
    int ksteps = K >> 5;
    int nt = tile / ksteps, ks = tile % ksteps;
    int n = nt * 16 + (lane & 15);
    int k = ks * 32 + ((lane & 16) ? 8 : 0) + (i & 7) + ((i & 8) ? 16 : 0);
    dst[idx] = f2bf(W[(size_t)n * K + k]);
}

// temp_hidden -> thcat[:, :256] (bf16), prev_hidden -> prev_b (bf16)
__global__ void __launch_bounds__(256)
convert_hidden_kernel(const float* __restrict__ temp, const float* __restrict__ prev,
                      unsigned short* __restrict__ thcat,
                      unsigned short* __restrict__ prev_b) {
    int idx = blockIdx.x * 256 + threadIdx.x;           // < NN*HH
    int n = idx >> 8, h = idx & 255;
    thcat[(size_t)n * 512 + h] = f2bf(temp[idx]);
    prev_b[idx] = f2bf(prev[idx]);
}

// te = temp_hidden @ Wt^T + bt   [NN,64]
__global__ void __launch_bounds__(256)
te_kernel(const unsigned short* __restrict__ thcat,
          const unsigned short* __restrict__ WtP,
          const float* __restrict__ bt, float* __restrict__ te) {
    int w = threadIdx.x >> 5, lane = threadIdx.x & 31;
    int row0 = blockIdx.x * 32 + (w >> 2) * 16;
    int ct = w & 3, c = (ct << 4) + (lane & 15);
    float bv = bt[c];
    v8f acc = {bv, bv, bv, bv, bv, bv, bv, bv};
    acc = gemm_tile(thcat + (size_t)row0 * 512, 512, WtP, ct, 8, lane, acc);
    int r0 = row0 + ((lane & 16) ? 8 : 0);
    for (int j = 0; j < 8; ++j) te[(size_t)(r0 + j) * 64 + c] = acc[j];
}

// Per-node: se GEMM (WMMA from LDS), softmax over 32 edges, Hagg, fused x_e.
__global__ void __launch_bounds__(256)
attn_kernel(const float* __restrict__ spat, const float* __restrict__ te,
            const float* __restrict__ xy, const float* __restrict__ W_xy,
            const float* __restrict__ bs, const unsigned short* __restrict__ WsP,
            unsigned short* __restrict__ thcat, unsigned short* __restrict__ cat) {
    __shared__ unsigned short sp[EE * HH];   // 32x256 bf16 tile of spat_hidden
    __shared__ float se[EE * AA];            // 32x64 f32
    __shared__ float wsm[EE];                // softmax weights
    int tid = threadIdx.x, n = blockIdx.x;

    // stage 1: global f32 -> LDS bf16 (stream once)
    const float4* s4 = (const float4*)(spat + (size_t)n * EE * HH);
    us4* d4 = (us4*)sp;
    for (int i = 0; i < 8; ++i) {
        int idx = tid + i * 256;
        float4 f = s4[idx];
        us4 h;
        h[0] = f2bf(f.x); h[1] = f2bf(f.y); h[2] = f2bf(f.z); h[3] = f2bf(f.w);
        d4[idx] = h;
    }
    __syncthreads();

    // stage 2: se[e,a] = spat[e,:]·Ws[a,:] + bs[a]   (2x4 tiles, 1 per wave)
    {
        int w = tid >> 5, lane = tid & 31;
        int m0 = (w >> 2) * 16, ct = w & 3, a0 = ct << 4;
        float bv = bs[a0 + (lane & 15)];
        v8f acc = {bv, bv, bv, bv, bv, bv, bv, bv};
        acc = gemm_tile(sp + m0 * HH, HH, WsP, ct, 8, lane, acc);
        int r0 = m0 + ((lane & 16) ? 8 : 0), c = a0 + (lane & 15);
        for (int j = 0; j < 8; ++j) se[(r0 + j) * AA + c] = acc[j];
    }
    __syncthreads();

    // stage 3: attn + softmax across 32 edges in wave 0
    if (tid < 32) {
        const float* tn = te + (size_t)n * AA;
        float s = 0.f;
        for (int a = 0; a < AA; ++a) s += se[tid * AA + a] * tn[a];
        s *= 4.0f;                               // E/sqrt(A) = 32/8
        float m = s;
        for (int off = 16; off > 0; off >>= 1) m = fmaxf(m, __shfl_xor(m, off, 32));
        float p = __expf(s - m);
        float sum = p;
        for (int off = 16; off > 0; off >>= 1) sum += __shfl_xor(sum, off, 32);
        wsm[tid] = p / sum;
    }
    __syncthreads();

    // stage 4: Hagg[h] = sum_e w[e]*spat[e,h]; fused x_e = relu(xy @ W_xy^T)
    {
        int h = tid;
        float acc = 0.f;
        for (int e = 0; e < EE; ++e) acc += wsm[e] * bf2f(sp[e * HH + h]);
        thcat[(size_t)n * 512 + 256 + h] = f2bf(acc);
        float x0 = xy[(size_t)n * 2], x1 = xy[(size_t)n * 2 + 1];
        float xe = fmaxf(x0 * W_xy[h * 2] + x1 * W_xy[h * 2 + 1], 0.f);
        cat[(size_t)n * 512 + h] = f2bf(xe);
    }
}

// H_e = relu([temp|Hagg] @ W_he^T) -> cat[:,256:512]
__global__ void __launch_bounds__(256)
he_kernel(const unsigned short* __restrict__ thcat,
          const unsigned short* __restrict__ WheP,
          unsigned short* __restrict__ cat) {
    int w = threadIdx.x >> 5, lane = threadIdx.x & 31;
    int row0 = blockIdx.x * 32 + (w >> 2) * 16;
    int ct = (blockIdx.y << 2) + (w & 3), col0 = ct << 4;
    v8f acc = {};
    acc = gemm_tile(thcat + (size_t)row0 * 512, 512, WheP, ct, 16, lane, acc);
    int r0 = row0 + ((lane & 16) ? 8 : 0), c = col0 + (lane & 15);
    for (int j = 0; j < 8; ++j)
        cat[(size_t)(r0 + j) * 512 + 256 + c] = f2bf(fmaxf(acc[j], 0.f));
}

// g=0: z = sigmoid(cat·Wih0 + prev·Whh0 + b0)
// g=1: rp = bf16(sigmoid(cat·Wih1 + prev·Whh1 + b1) * prev)
// g=2: g2a = cat·Wih2 + b2
__global__ void __launch_bounds__(256)
gates_kernel(const unsigned short* __restrict__ cat,
             const unsigned short* __restrict__ prev_b,
             const unsigned short* __restrict__ WihP,
             const unsigned short* __restrict__ WhhP,
             const float* __restrict__ bias_g, const float* __restrict__ prev,
             float* __restrict__ z_out, unsigned short* __restrict__ rp_out,
             float* __restrict__ g2a_out) {
    int w = threadIdx.x >> 5, lane = threadIdx.x & 31;
    int g = blockIdx.z;
    int row0 = blockIdx.x * 32 + (w >> 2) * 16;
    int ct = (blockIdx.y << 2) + (w & 3), col0 = ct << 4;
    v8f acc = {};
    acc = gemm_tile(cat + (size_t)row0 * 512, 512,
                    WihP + (size_t)g * (16 * 16 * 512), ct, 16, lane, acc);
    if (g < 2)
        acc = gemm_tile(prev_b + (size_t)row0 * 256, 256,
                        WhhP + (size_t)g * (16 * 8 * 512), ct, 8, lane, acc);
    int r0 = row0 + ((lane & 16) ? 8 : 0), c = col0 + (lane & 15);
    float b = bias_g[g * 256 + c];
    for (int j = 0; j < 8; ++j) {
        size_t idx = (size_t)(r0 + j) * 256 + c;
        float val = acc[j] + b;
        if (g == 0) {
            z_out[idx] = 1.f / (1.f + __expf(-val));
        } else if (g == 1) {
            float rv = 1.f / (1.f + __expf(-val));
            rp_out[idx] = f2bf(rv * prev[idx]);
        } else {
            g2a_out[idx] = val;
        }
    }
}

// n = relu(g2a + rp·Whh2); hidden = (1-z)*n + z*prev  (written into d_out)
__global__ void __launch_bounds__(256)
hidden_kernel(const unsigned short* __restrict__ rp,
              const unsigned short* __restrict__ WhhP,
              const float* __restrict__ g2a, const float* __restrict__ z,
              const float* __restrict__ prev, float* __restrict__ hid) {
    int w = threadIdx.x >> 5, lane = threadIdx.x & 31;
    int row0 = blockIdx.x * 32 + (w >> 2) * 16;
    int ct = (blockIdx.y << 2) + (w & 3), col0 = ct << 4;
    v8f acc = {};
    acc = gemm_tile(rp + (size_t)row0 * 256, 256,
                    WhhP + (size_t)2 * (16 * 8 * 512), ct, 8, lane, acc);
    int r0 = row0 + ((lane & 16) ? 8 : 0), c = col0 + (lane & 15);
    for (int j = 0; j < 8; ++j) {
        size_t idx = (size_t)(r0 + j) * 256 + c;
        float nv = fmaxf(acc[j] + g2a[idx], 0.f);
        float zv = z[idx];
        hid[idx] = (1.f - zv) * nv + zv * prev[idx];
    }
}

// predict = hidden @ W_pred^T + b_pred   (one wave per node, 5 dots of 256)
__global__ void __launch_bounds__(256)
pred_kernel(const float* __restrict__ hid, const float* __restrict__ W_pred,
            const float* __restrict__ b_pred, float* __restrict__ pred) {
    int w = threadIdx.x >> 5, lane = threadIdx.x & 31;
    int n = blockIdx.x * 8 + w;
    const float* hd = hid + (size_t)n * HH;
    float s0 = 0, s1 = 0, s2 = 0, s3 = 0, s4 = 0;
    for (int h = lane; h < HH; h += 32) {
        float hv = hd[h];
        s0 += hv * W_pred[0 * HH + h];
        s1 += hv * W_pred[1 * HH + h];
        s2 += hv * W_pred[2 * HH + h];
        s3 += hv * W_pred[3 * HH + h];
        s4 += hv * W_pred[4 * HH + h];
    }
    for (int off = 16; off > 0; off >>= 1) {
        s0 += __shfl_xor(s0, off, 32); s1 += __shfl_xor(s1, off, 32);
        s2 += __shfl_xor(s2, off, 32); s3 += __shfl_xor(s3, off, 32);
        s4 += __shfl_xor(s4, off, 32);
    }
    if (lane == 0) {
        float* o = pred + (size_t)n * ODIM;
        o[0] = s0 + b_pred[0]; o[1] = s1 + b_pred[1]; o[2] = s2 + b_pred[2];
        o[3] = s3 + b_pred[3]; o[4] = s4 + b_pred[4];
    }
}

// ---------------------------------------------------------------------------
extern "C" void kernel_launch(void* const* d_in, const int* in_sizes, int n_in,
                              void* d_out, int out_size, void* d_ws, size_t ws_size,
                              hipStream_t stream) {
    const float* xy        = (const float*)d_in[0];
    const float* temp      = (const float*)d_in[1];
    const float* spat      = (const float*)d_in[2];
    const float* prev      = (const float*)d_in[3];
    const float* Wt        = (const float*)d_in[4];
    const float* bt        = (const float*)d_in[5];
    const float* Ws        = (const float*)d_in[6];
    const float* bs        = (const float*)d_in[7];
    const float* W_xy      = (const float*)d_in[8];
    const float* W_he      = (const float*)d_in[9];
    const float* weight_ih = (const float*)d_in[10];
    const float* weight_hh = (const float*)d_in[11];
    const float* bias_g    = (const float*)d_in[12];
    const float* W_pred    = (const float*)d_in[13];
    const float* b_pred    = (const float*)d_in[14];

    float* out_pred   = (float*)d_out;
    float* out_hidden = out_pred + (size_t)NN * ODIM;

    // --- carve workspace (all offsets 256B aligned) ---
    char* p = (char*)d_ws;
    auto carve = [&](size_t bytes) {
        void* r = (void*)p;
        p += (bytes + 255) & ~(size_t)255;
        return r;
    };
    unsigned short* WtP   = (unsigned short*)carve((size_t)4 * 8 * 512 * 2);
    unsigned short* WsP   = (unsigned short*)carve((size_t)4 * 8 * 512 * 2);
    unsigned short* WheP  = (unsigned short*)carve((size_t)16 * 16 * 512 * 2);
    unsigned short* WihP  = (unsigned short*)carve((size_t)48 * 16 * 512 * 2);
    unsigned short* WhhP  = (unsigned short*)carve((size_t)48 * 8 * 512 * 2);
    float*          te    = (float*)carve((size_t)NN * AA * 4);
    unsigned short* thcat = (unsigned short*)carve((size_t)NN * 512 * 2);
    unsigned short* cat   = (unsigned short*)carve((size_t)NN * 512 * 2);
    unsigned short* prevb = (unsigned short*)carve((size_t)NN * HH * 2);
    float*          zbuf  = (float*)carve((size_t)NN * HH * 4);
    unsigned short* rp    = (unsigned short*)carve((size_t)NN * HH * 2);
    float*          g2a   = (float*)carve((size_t)NN * HH * 4);
    (void)ws_size; (void)in_sizes; (void)n_in; (void)out_size;

    // --- weight packing (bf16 WMMA B-fragments) ---
    pack_b_kernel<<<64,   256, 0, stream>>>(Wt,        WtP,  64,  256);
    pack_b_kernel<<<64,   256, 0, stream>>>(Ws,        WsP,  64,  256);
    pack_b_kernel<<<512,  256, 0, stream>>>(W_he,      WheP, 256, 512);
    pack_b_kernel<<<1536, 256, 0, stream>>>(weight_ih, WihP, 768, 512);
    pack_b_kernel<<<768,  256, 0, stream>>>(weight_hh, WhhP, 768, 256);

    // --- activation conversion ---
    convert_hidden_kernel<<<NN * HH / 256, 256, 0, stream>>>(temp, prev, thcat, prevb);

    // --- pipeline ---
    te_kernel<<<NN / 32, 256, 0, stream>>>(thcat, WtP, bt, te);
    attn_kernel<<<NN, 256, 0, stream>>>(spat, te, xy, W_xy, bs, WsP, thcat, cat);
    he_kernel<<<dim3(NN / 32, 4), 256, 0, stream>>>(thcat, WheP, cat);
    gates_kernel<<<dim3(NN / 32, 4, 3), 256, 0, stream>>>(cat, prevb, WihP, WhhP,
                                                          bias_g, prev,
                                                          zbuf, rp, g2a);
    hidden_kernel<<<dim3(NN / 32, 4), 256, 0, stream>>>(rp, WhhP, g2a, zbuf,
                                                        prev, out_hidden);
    pred_kernel<<<NN / 8, 256, 0, stream>>>(out_hidden, W_pred, b_pred, out_pred);
}
